// AMIPRouterInference_41815801594568
// MI455X (gfx1250) — compile-verified, matrix operand order checked by reading, not run on hardware
//
#include <hip/hip_runtime.h>
#include <math.h>

// ============================================================================
// AMIPRouterInference for MI455X (gfx1250, wave32, WMMA f32_16x16x32_f16)
//
// Factorized algorithm (vs. 2.06 TFLOP naive):
//   anch_proj[e] = h_L @ W1[e][:D]              (WMMA GEMM, 34 GFLOP)
//   base[e]      = h_L @ W1[e][D:] + b1[e]      (WMMA GEMM, 34 GFLOP)
//   weights      = softmax(h_L @ W_route + b_route)
//   q = h_L@Wq+bq ; kproj = h_L@Wk+bk           (WMMA GEMM, 2 GFLOP)
//   w,S,keep     = neighbor softmax over 2R=20 offsets (VALU)
//   mixw[m,e,:]  = wt_e * sum_r w_r * gelu(anch_proj[e][posc_r]+base[e][m])
//                  (elementwise, ~660MB of L2-resident reads)
//   out          = mixw[1024,8192] @ W2cat[8192,2048]   (WMMA GEMM, 34 GFLOP)
//   out          = keep ? out + S * sum_e wt_e*b2[e] : 0
//
// GEMM uses CDNA5 async global->LDS copies (ASYNCcnt) + double-buffered LDS
// (one barrier per BK step) when the builtins exist; otherwise falls back to
// the proven VGPR-staged single-buffer pipeline.
// ============================================================================

typedef __attribute__((ext_vector_type(16))) _Float16 v16h;
typedef __attribute__((ext_vector_type(8)))  float    v8f;
typedef __attribute__((ext_vector_type(4)))  float    fvec4;
typedef __attribute__((ext_vector_type(4)))  int      v4i;

#define Bb  2
#define Ll  512
#define Dd  2048
#define Kk  8
#define Rr  10
#define NR  20
#define DH  1024
#define Pp  256
#define Mm  1024   // B*L tokens

// GEMM tiling: 256 threads = 8 wave32 waves, waves in 4x2 grid of 32x32 tiles
#define BM  128
#define BN  64
#define BK  32
#define BKP 40     // padded LDS pitch in halves (conflict-free fragment loads)

#ifndef __has_builtin
#define __has_builtin(x) 0
#endif

#if __has_builtin(__builtin_amdgcn_global_load_async_to_lds_b128)
#define USE_ASYNC_COPY 1
// prototype: (v4i addrspace(1)*, v4i addrspace(3)*, imm offset, imm cpol)
#define ASYNC_LOAD_B128(g, l)                                          \
  __builtin_amdgcn_global_load_async_to_lds_b128(                      \
      (__attribute__((address_space(1))) v4i*)(g),                     \
      (__attribute__((address_space(3))) v4i*)(l), 0, 0)
#else
#define USE_ASYNC_COPY 0
#endif

#if __has_builtin(__builtin_amdgcn_s_wait_asynccnt)
#define WAIT_ASYNC0() __builtin_amdgcn_s_wait_asynccnt(0)
#else
#define WAIT_ASYNC0() asm volatile("s_wait_asynccnt 0" ::: "memory")
#endif

union FragU { v16h v; fvec4 f[2]; };

// ---------------------------------------------------------------------------
// fp32 -> f16 elementwise convert
// ---------------------------------------------------------------------------
__global__ void cvt_f32_to_f16(const float* __restrict__ in,
                               _Float16* __restrict__ out, int n) {
  int i = blockIdx.x * blockDim.x + threadIdx.x;
  if (i < n) out[i] = (_Float16)in[i];
}

// ---------------------------------------------------------------------------
// fp32 (rows x cols, row-major) -> f16 transposed (cols x rows, row-major)
// 32x32 tile through LDS, block = 256 threads (32x8)
// ---------------------------------------------------------------------------
__global__ __launch_bounds__(256) void tcvt_kernel(
    const float* __restrict__ in, _Float16* __restrict__ out,
    int rows, int cols, long long ibs, long long obs) {
  __shared__ float tile[32][33];
  in  += (long long)blockIdx.z * ibs;
  out += (long long)blockIdx.z * obs;
  int tx = threadIdx.x & 31, ty = threadIdx.x >> 5;
  int c = blockIdx.x * 32 + tx;
#pragma unroll
  for (int i = 0; i < 32; i += 8) {
    int r = blockIdx.y * 32 + ty + i;
    tile[ty + i][tx] = in[(long long)r * cols + c];
  }
  __syncthreads();
  int oc = blockIdx.y * 32 + tx;  // out col = original row
#pragma unroll
  for (int i = 0; i < 32; i += 8) {
    int orow = blockIdx.x * 32 + ty + i;  // out row = original col
    out[(long long)orow * rows + oc] = (_Float16)tile[tx][ty + i];
  }
}

// ---------------------------------------------------------------------------
// Generic f16 WMMA GEMM: C[M,N](f32) = A[M,Kd](f16, lda) * B^T-layout
// B is stored transposed: Bm[n*ldb + k]. Batched over blockIdx.z.
// grid = (N/BN, M/BM, batch); block = 256.
// ---------------------------------------------------------------------------
__global__ __launch_bounds__(256) void gemm_wmma_f16(
    const _Float16* __restrict__ A, int lda, long long a_bs,
    const _Float16* __restrict__ Bm, int ldb, long long b_bs,
    const float* __restrict__ bias, int bias_bs,
    float* __restrict__ C, int ldc, long long c_bs, int Kd) {
  const int z = blockIdx.z;
  A  += (long long)z * a_bs;
  Bm += (long long)z * b_bs;
  C  += (long long)z * c_bs;

  const int m0 = blockIdx.y * BM;
  const int n0 = blockIdx.x * BN;
  const int t    = threadIdx.x;
  const int lane = t & 31;
  const int wv   = t >> 5;
  const int wm0  = (wv >> 1) << 5;   // 0,32,64,96
  const int wn0  = (wv & 1) << 5;    // 0,32
  const int lm   = lane & 15;
  const int ksel = (lane >> 4) << 3; // 0 or 8 halves (lane-group K offset)

  v8f c00 = {0.f,0.f,0.f,0.f,0.f,0.f,0.f,0.f};
  v8f c01 = c00, c10 = c00, c11 = c00;

  // Per-thread staging coordinates (16B per thread per chunk).
  const int rA0 = t >> 2;            // A rows 0..63   (pass 0)
  const int rA1 = 64 + rA0;          // A rows 64..127 (pass 1)
  const int rB  = t >> 2;            // B rows 0..63
  const int ch  = (t & 3) << 3;      // half offset 0,8,16,24
  const _Float16* gA0 = A  + (long long)(m0 + rA0) * lda + ch;
  const _Float16* gA1 = A  + (long long)(m0 + rA1) * lda + ch;
  const _Float16* gB  = Bm + (long long)(n0 + rB)  * ldb + ch;
  const int la0 = rA0 * BKP + ch;
  const int la1 = rA1 * BKP + ch;
  const int lb  = rB  * BKP + ch;

#if USE_ASYNC_COPY
  // ---- CDNA5 path: async global->LDS, double-buffered, 1 barrier/step ----
  __shared__ _Float16 sA[2][BM * BKP];
  __shared__ _Float16 sB[2][BN * BKP];

  // prologue: stage tile 0 into buffer 0
  ASYNC_LOAD_B128(gA0, &sA[0][la0]);
  ASYNC_LOAD_B128(gA1, &sA[0][la1]);
  ASYNC_LOAD_B128(gB,  &sB[0][lb]);

  const int nsteps = Kd / BK;
  for (int i = 0; i < nsteps; ++i) {
    const int cur = i & 1;
    WAIT_ASYNC0();        // this wave's loads for tile i are in LDS
    __syncthreads();      // all waves' loads done; all reads of tile i-1 done
    if (i + 1 < nsteps) { // safe to overwrite buffer cur^1 now
      const long long koff = (long long)(i + 1) * BK;
      ASYNC_LOAD_B128(gA0 + koff, &sA[cur ^ 1][la0]);
      ASYNC_LOAD_B128(gA1 + koff, &sA[cur ^ 1][la1]);
      ASYNC_LOAD_B128(gB  + koff, &sB[cur ^ 1][lb]);
    }
    // Per-lane fragments per ISA 16-bit A/B layout:
    //   lane L (group g=L>>4): M/N = L&15, K halves {8g..8g+7, 8g+16..8g+23}
    FragU a0, a1, b0, b1;
    const _Float16* p;
    p = &sA[cur][(wm0 + lm)      * BKP + ksel]; a0.f[0] = *(const fvec4*)p; a0.f[1] = *(const fvec4*)(p + 16);
    p = &sA[cur][(wm0 + 16 + lm) * BKP + ksel]; a1.f[0] = *(const fvec4*)p; a1.f[1] = *(const fvec4*)(p + 16);
    p = &sB[cur][(wn0 + lm)      * BKP + ksel]; b0.f[0] = *(const fvec4*)p; b0.f[1] = *(const fvec4*)(p + 16);
    p = &sB[cur][(wn0 + 16 + lm) * BKP + ksel]; b1.f[0] = *(const fvec4*)p; b1.f[1] = *(const fvec4*)(p + 16);

    c00 = __builtin_amdgcn_wmma_f32_16x16x32_f16(false, a0.v, false, b0.v, (short)0, c00, false, false);
    c01 = __builtin_amdgcn_wmma_f32_16x16x32_f16(false, a0.v, false, b1.v, (short)0, c01, false, false);
    c10 = __builtin_amdgcn_wmma_f32_16x16x32_f16(false, a1.v, false, b0.v, (short)0, c10, false, false);
    c11 = __builtin_amdgcn_wmma_f32_16x16x32_f16(false, a1.v, false, b1.v, (short)0, c11, false, false);
  }
#else
  // ---- Fallback: VGPR-staged single buffer (2 barriers/step) ----
  __shared__ _Float16 sA1[BM * BKP];
  __shared__ _Float16 sB1[BN * BKP];
  for (int k0 = 0; k0 < Kd; k0 += BK) {
    *(fvec4*)&sA1[la0] = *(const fvec4*)(gA0 + k0);
    *(fvec4*)&sA1[la1] = *(const fvec4*)(gA1 + k0);
    *(fvec4*)&sB1[lb]  = *(const fvec4*)(gB  + k0);
    __syncthreads();
    FragU a0, a1, b0, b1;
    const _Float16* p;
    p = &sA1[(wm0 + lm)      * BKP + ksel]; a0.f[0] = *(const fvec4*)p; a0.f[1] = *(const fvec4*)(p + 16);
    p = &sA1[(wm0 + 16 + lm) * BKP + ksel]; a1.f[0] = *(const fvec4*)p; a1.f[1] = *(const fvec4*)(p + 16);
    p = &sB1[(wn0 + lm)      * BKP + ksel]; b0.f[0] = *(const fvec4*)p; b0.f[1] = *(const fvec4*)(p + 16);
    p = &sB1[(wn0 + 16 + lm) * BKP + ksel]; b1.f[0] = *(const fvec4*)p; b1.f[1] = *(const fvec4*)(p + 16);

    c00 = __builtin_amdgcn_wmma_f32_16x16x32_f16(false, a0.v, false, b0.v, (short)0, c00, false, false);
    c01 = __builtin_amdgcn_wmma_f32_16x16x32_f16(false, a0.v, false, b1.v, (short)0, c01, false, false);
    c10 = __builtin_amdgcn_wmma_f32_16x16x32_f16(false, a1.v, false, b0.v, (short)0, c10, false, false);
    c11 = __builtin_amdgcn_wmma_f32_16x16x32_f16(false, a1.v, false, b1.v, (short)0, c11, false, false);
    __syncthreads();
  }
#endif

  // Epilogue per C/D layout: VGPR i -> M = i + 8*(lane>=16), N = lane&15
  const int cn   = lane & 15;
  const int rsel = (lane >> 4) << 3;
  {
    int col = n0 + wn0 + cn;
    float bv = bias ? bias[(long long)z * bias_bs + col] : 0.f;
#pragma unroll
    for (int i = 0; i < 8; ++i) {
      C[(long long)(m0 + wm0 + rsel + i) * ldc + col]      = c00[i] + bv;
      C[(long long)(m0 + wm0 + 16 + rsel + i) * ldc + col] = c10[i] + bv;
    }
  }
  {
    int col = n0 + wn0 + 16 + cn;
    float bv = bias ? bias[(long long)z * bias_bs + col] : 0.f;
#pragma unroll
    for (int i = 0; i < 8; ++i) {
      C[(long long)(m0 + wm0 + rsel + i) * ldc + col]      = c01[i] + bv;
      C[(long long)(m0 + wm0 + 16 + rsel + i) * ldc + col] = c11[i] + bv;
    }
  }
}

// ---------------------------------------------------------------------------
// Routing softmax: weights[m,:] = softmax(h_L[m] @ W_route + b_route)
// one block (256 thr) per token
// ---------------------------------------------------------------------------
__global__ __launch_bounds__(256) void route_kernel(
    const float* __restrict__ h, const float* __restrict__ Wr,
    const float* __restrict__ br, float* __restrict__ wts) {
  int m = blockIdx.x, t = threadIdx.x;
  float acc[Kk];
#pragma unroll
  for (int e = 0; e < Kk; ++e) acc[e] = 0.f;
  const float* hr = h + (long long)m * Dd;
  for (int d = t; d < Dd; d += 256) {
    float hv = hr[d];
    const float* wrow = Wr + (long long)d * Kk;
#pragma unroll
    for (int e = 0; e < Kk; ++e) acc[e] += hv * wrow[e];
  }
  __shared__ float red[256];
  __shared__ float logit[Kk];
  for (int e = 0; e < Kk; ++e) {
    red[t] = acc[e];
    __syncthreads();
    for (int s = 128; s > 0; s >>= 1) {
      if (t < s) red[t] += red[t + s];
      __syncthreads();
    }
    if (t == 0) logit[e] = red[0] + br[e];
    __syncthreads();
  }
  if (t == 0) {
    float mx = -1e30f;
#pragma unroll
    for (int e = 0; e < Kk; ++e) mx = fmaxf(mx, logit[e]);
    float den = 0.f, ex[Kk];
#pragma unroll
    for (int e = 0; e < Kk; ++e) { ex[e] = expf(logit[e] - mx); den += ex[e]; }
#pragma unroll
    for (int e = 0; e < Kk; ++e) wts[m * Kk + e] = ex[e] / den;
  }
}

// ---------------------------------------------------------------------------
// Neighbor attention: scores, validity, softmax -> w[m,20], S[m], keep[m]
// one block (256 thr) per token (P=256 lanes == one element each)
// ---------------------------------------------------------------------------
__global__ __launch_bounds__(256) void attn_kernel(
    const float* __restrict__ q, const float* __restrict__ kp,
    const unsigned char* __restrict__ masked,
    float* __restrict__ w_out, float* __restrict__ S_out,
    int* __restrict__ keep_out) {
  int m = blockIdx.x, t = threadIdx.x;
  int b = m >> 9, l = m & (Ll - 1);
  __shared__ float red[256];
  __shared__ float sc[NR];
  float qv = q[(long long)m * Pp + t];
  for (int r = 0; r < NR; ++r) {
    int off  = (r < Rr) ? (r - Rr) : (r - Rr + 1);
    int pos  = l + off;
    int posc = pos < 0 ? 0 : (pos > Ll - 1 ? Ll - 1 : pos);
    float kv = kp[((long long)b * Ll + posc) * Pp + t];
    red[t] = qv * kv;
    __syncthreads();
    for (int s = 128; s > 0; s >>= 1) {
      if (t < s) red[t] += red[t + s];
      __syncthreads();
    }
    if (t == 0) sc[r] = red[0] * 0.0625f;  // / sqrt(256)
    __syncthreads();
  }
  if (t == 0) {
    float sv[NR];
    int   vd[NR];
    int any = 0;
    float mx = -1e30f;
#pragma unroll
    for (int r = 0; r < NR; ++r) {
      int off  = (r < Rr) ? (r - Rr) : (r - Rr + 1);
      int pos  = l + off;
      int inb  = (pos >= 0) && (pos < Ll);
      int posc = pos < 0 ? 0 : (pos > Ll - 1 ? Ll - 1 : pos);
      vd[r] = inb && (masked[b * Ll + posc] == 0);
      any |= vd[r];
      sv[r] = vd[r] ? sc[r] : -1e9f;
      mx = fmaxf(mx, sv[r]);
    }
    float den = 0.f, ex[NR];
#pragma unroll
    for (int r = 0; r < NR; ++r) { ex[r] = expf(sv[r] - mx); den += ex[r]; }
    float S = 0.f;
#pragma unroll
    for (int r = 0; r < NR; ++r) {
      float w = vd[r] ? ex[r] / den : 0.f;
      w_out[m * NR + r] = w;
      S += w;
    }
    S_out[m] = S;
    keep_out[m] = (masked[m] != 0) && any;
  }
}

// ---------------------------------------------------------------------------
// Mix: mixw[m, e*DH+j] = wt_e * sum_r w_r * gelu(anch[e][posc_r][j]+base[e][m][j])
// grid (M, K); 256 threads, 4 elements each (DH=1024). Source data L2-resident.
// ---------------------------------------------------------------------------
__global__ __launch_bounds__(256) void mix_kernel(
    const float* __restrict__ anch, const float* __restrict__ base,
    const float* __restrict__ w_in, const float* __restrict__ wts,
    const int* __restrict__ keep, _Float16* __restrict__ mixw) {
  int m = blockIdx.x, e = blockIdx.y, t = threadIdx.x;
  int b = m >> 9, l = m & (Ll - 1);
  float acc[4] = {0.f, 0.f, 0.f, 0.f};
  if (keep[m]) {
    const float* bp = base + ((long long)e * Mm + m) * DH;
    float bs[4];
#pragma unroll
    for (int i = 0; i < 4; ++i) bs[i] = bp[t + i * 256];
    for (int r = 0; r < NR; ++r) {
      float wr = w_in[m * NR + r];
      if (wr == 0.f) continue;
      int off  = (r < Rr) ? (r - Rr) : (r - Rr + 1);
      int pos  = l + off;
      int posc = pos < 0 ? 0 : (pos > Ll - 1 ? Ll - 1 : pos);
      const float* ap = anch + ((long long)e * Mm + (long long)b * Ll + posc) * DH;
#pragma unroll
      for (int i = 0; i < 4; ++i) {
        float x = ap[t + i * 256] + bs[i];
        float g = 0.5f * x * (1.f + erff(x * 0.70710678118654752f));
        acc[i] += wr * g;
      }
    }
    float wt = wts[m * Kk + e];
#pragma unroll
    for (int i = 0; i < 4; ++i) acc[i] *= wt;
  }
#pragma unroll
  for (int i = 0; i < 4; ++i)
    mixw[(long long)m * (Kk * DH) + e * DH + t + i * 256] = (_Float16)acc[i];
}

// ---------------------------------------------------------------------------
// Epilogue: out = keep ? out + S * sum_e wt_e*b2[e][d] : 0   (in-place on C)
// ---------------------------------------------------------------------------
__global__ void epilogue_kernel(float* __restrict__ out,
                                const int* __restrict__ keep,
                                const float* __restrict__ wts,
                                const float* __restrict__ b2,
                                const float* __restrict__ S) {
  int idx = blockIdx.x * blockDim.x + threadIdx.x;
  int m = idx >> 11;        // / 2048
  int d = idx & (Dd - 1);
  float v = 0.f;
  if (keep[m]) {
    float bsum = 0.f;
#pragma unroll
    for (int e = 0; e < Kk; ++e) bsum += wts[m * Kk + e] * b2[e * Dd + d];
    v = out[idx] + S[m] * bsum;
  }
  out[idx] = v;
}

// ---------------------------------------------------------------------------
extern "C" void kernel_launch(void* const* d_in, const int* in_sizes, int n_in,
                              void* d_out, int out_size, void* d_ws, size_t ws_size,
                              hipStream_t stream) {
  (void)in_sizes; (void)n_in; (void)out_size; (void)ws_size;
  const float* h_L     = (const float*)d_in[0];
  const unsigned char* masked = (const unsigned char*)d_in[1];  // jax bool = 1 byte
  const float* W_route = (const float*)d_in[2];
  const float* b_route = (const float*)d_in[3];
  const float* W1      = (const float*)d_in[4];
  const float* b1      = (const float*)d_in[5];
  const float* W2      = (const float*)d_in[6];
  const float* b2      = (const float*)d_in[7];
  const float* Wq      = (const float*)d_in[8];
  const float* bq      = (const float*)d_in[9];
  const float* Wk      = (const float*)d_in[10];
  const float* bk      = (const float*)d_in[11];
  float* out = (float*)d_out;

  char* ws = (char*)d_ws;
  size_t off = 0;
  auto alloc = [&](size_t bytes) -> char* {
    char* r = ws + off;
    off = (off + bytes + 255) & ~(size_t)255;
    return r;
  };

  _Float16* h16  = (_Float16*)alloc((size_t)Mm * Dd * 2);          // 4 MB
  _Float16* W1T  = (_Float16*)alloc((size_t)Kk * DH * 2 * Dd * 2); // 67 MB
  _Float16* W2T  = (_Float16*)alloc((size_t)Dd * Kk * DH * 2);     // 33.5 MB
  _Float16* WqT  = (_Float16*)alloc((size_t)Pp * Dd * 2);
  _Float16* WkT  = (_Float16*)alloc((size_t)Pp * Dd * 2);
  float* anchb   = (float*)alloc((size_t)Kk * Mm * DH * 4);        // 33.5 MB
  float* baseb   = (float*)alloc((size_t)Kk * Mm * DH * 4);        // 33.5 MB
  float* qbuf    = (float*)alloc((size_t)Mm * Pp * 4);
  float* kbuf    = (float*)alloc((size_t)Mm * Pp * 4);
  float* wtsb    = (float*)alloc((size_t)Mm * Kk * 4);
  float* wsm     = (float*)alloc((size_t)Mm * NR * 4);
  float* Ssum    = (float*)alloc((size_t)Mm * 4);
  int*   keepb   = (int*)alloc((size_t)Mm * 4);
  _Float16* mixw = (_Float16*)alloc((size_t)Mm * Kk * DH * 2);     // 16.7 MB
  // total ~193 MB of workspace

  // 1) conversions / transposes of weights (once per launch, ~250MB HBM)
  cvt_f32_to_f16<<<(Mm * Dd) / 256, 256, 0, stream>>>(h_L, h16, Mm * Dd);
  tcvt_kernel<<<dim3(DH / 32, (2 * Dd) / 32, Kk), 256, 0, stream>>>(
      W1, W1T, 2 * Dd, DH, (long long)2 * Dd * DH, (long long)2 * Dd * DH);
  tcvt_kernel<<<dim3(Dd / 32, (Kk * DH) / 32, 1), 256, 0, stream>>>(
      W2, W2T, Kk * DH, Dd, 0, 0);
  tcvt_kernel<<<dim3(Pp / 32, Dd / 32, 1), 256, 0, stream>>>(Wq, WqT, Dd, Pp, 0, 0);
  tcvt_kernel<<<dim3(Pp / 32, Dd / 32, 1), 256, 0, stream>>>(Wk, WkT, Dd, Pp, 0, 0);

  // 2) routing softmax
  route_kernel<<<Mm, 256, 0, stream>>>(h_L, W_route, b_route, wtsb);

  // 3) per-token expert projections (batched over e), WMMA
  //    anch_proj[e] = h16 @ W1[e][:D]   (W1T_e rows n, cols k in [0,2048))
  gemm_wmma_f16<<<dim3(DH / BN, Mm / BM, Kk), 256, 0, stream>>>(
      h16, Dd, 0,
      W1T, 2 * Dd, (long long)DH * 2 * Dd,
      nullptr, 0,
      anchb, DH, (long long)Mm * DH, Dd);
  //    base[e] = h16 @ W1[e][D:] + b1[e]  (k offset 2048 into W1T rows)
  gemm_wmma_f16<<<dim3(DH / BN, Mm / BM, Kk), 256, 0, stream>>>(
      h16, Dd, 0,
      W1T + Dd, 2 * Dd, (long long)DH * 2 * Dd,
      b1, DH,
      baseb, DH, (long long)Mm * DH, Dd);
  //    q / kproj
  gemm_wmma_f16<<<dim3(Pp / BN, Mm / BM, 1), 256, 0, stream>>>(
      h16, Dd, 0, WqT, Dd, 0, bq, 0, qbuf, Pp, 0, Dd);
  gemm_wmma_f16<<<dim3(Pp / BN, Mm / BM, 1), 256, 0, stream>>>(
      h16, Dd, 0, WkT, Dd, 0, bk, 0, kbuf, Pp, 0, Dd);

  // 4) neighbor attention softmax (w, S, keep)
  attn_kernel<<<Mm, 256, 0, stream>>>(qbuf, kbuf, masked, wsm, Ssum, keepb);

  // 5) gelu + neighbor mix (weight-folded), f16 output for final GEMM
  mix_kernel<<<dim3(Mm, Kk), 256, 0, stream>>>(anchb, baseb, wsm, wtsb, keepb, mixw);

  // 6) single big GEMM: out[1024,2048] = mixw[1024,8192] @ W2cat[8192,2048]
  gemm_wmma_f16<<<dim3(Dd / BN, Mm / BM, 1), 256, 0, stream>>>(
      mixw, Kk * DH, 0, W2T, Kk * DH, 0, nullptr, 0, out, Dd, 0, Kk * DH);

  // 7) bias + keep mask (in place)
  epilogue_kernel<<<(Mm * Dd) / 256, 256, 0, stream>>>(out, keepb, wtsb, b2, Ssum);
}
